// BitAttention_78245714199347
// MI455X (gfx1250) — 8-bit matrix operand order repaired and checked by reading, compiled, not run on hardware
//
#include <hip/hip_runtime.h>
#include <stdint.h>

typedef __attribute__((ext_vector_type(16))) _Float16 v16h;
typedef __attribute__((ext_vector_type(8)))  int      v8i;
typedef __attribute__((ext_vector_type(8)))  float    v8f;

#define D_MODEL 1024
#define N_TOK   4096
#define SEQ_T   2048
#define NH      16
#define DH      64
#define ATT_SCALE 0.125f
#define W_ELEMS 1048576

__device__ __forceinline__ v8i zero8i(){ v8i z;
#pragma unroll
  for(int i=0;i<8;++i) z[i]=0; return z; }
__device__ __forceinline__ v8f zero8f(){ v8f z;
#pragma unroll
  for(int i=0;i<8;++i) z[i]=0.f; return z; }

__device__ __forceinline__ float redmax16(float v){
  v = fmaxf(v, __shfl_xor(v,1,32));
  v = fmaxf(v, __shfl_xor(v,2,32));
  v = fmaxf(v, __shfl_xor(v,4,32));
  v = fmaxf(v, __shfl_xor(v,8,32));
  return v;
}
__device__ __forceinline__ float redsum16(float v){
  v += __shfl_xor(v,1,32);
  v += __shfl_xor(v,2,32);
  v += __shfl_xor(v,4,32);
  v += __shfl_xor(v,8,32);
  return v;
}

__device__ __forceinline__ uint32_t lds_off(void* p){
  return (uint32_t)(uintptr_t)(__attribute__((address_space(3))) char*)p;
}
__device__ __forceinline__ void async_copy_b128(uint32_t lds_byte, uint64_t gaddr){
  asm volatile("global_load_async_to_lds_b128 %0, %1, off"
               :: "v"(lds_byte), "v"(gaddr) : "memory");
}

__device__ __forceinline__ v8i load_a8(const int8_t* row, int half){
  const int2 p0 = *(const int2*)(row +      half*8);
  const int2 p1 = *(const int2*)(row + 16 + half*8);
  const int2 p2 = *(const int2*)(row + 32 + half*8);
  const int2 p3 = *(const int2*)(row + 48 + half*8);
  v8i r; r[0]=p0.x; r[1]=p0.y; r[2]=p1.x; r[3]=p1.y;
         r[4]=p2.x; r[5]=p2.y; r[6]=p3.x; r[7]=p3.y;
  return r;
}
__device__ __forceinline__ v8i load_b8(const int8_t* row, int half){
  const int2 p0 = *(const int2*)(row +      half*8);
  const int2 p1 = *(const int2*)(row + 16 + half*8);
  const int2 p2 = *(const int2*)(row + 32 + half*8);
  const int2 p3 = *(const int2*)(row + 48 + half*8);
  v8i r; r[0]=p0.x; r[1]=p0.y; r[2]=p1.x; r[3]=p1.y;
         r[4]=p2.x; r[5]=p2.y; r[6]=p3.x; r[7]=p3.y;
  return r;
}
__device__ __forceinline__ v16h load_h16f(const _Float16* p, int half){
  const int4 x0 = *(const int4*)(p +      half*8);
  const int4 x1 = *(const int4*)(p + 16 + half*8);
  union { int i[8]; v16h h; } u;
  u.i[0]=x0.x; u.i[1]=x0.y; u.i[2]=x0.z; u.i[3]=x0.w;
  u.i[4]=x1.x; u.i[5]=x1.y; u.i[6]=x1.z; u.i[7]=x1.w;
  return u.h;
}

__global__ void init_zero_kernel(float* acc, float* ent){
  if (threadIdx.x < 4) acc[threadIdx.x] = 0.f;
  if (threadIdx.x == 0) *ent = 0.f;
}

__global__ void __launch_bounds__(256)
wabs_sum_kernel(const float* __restrict__ w0, const float* __restrict__ w1,
                const float* __restrict__ w2, const float* __restrict__ w3,
                float* __restrict__ acc){
  __shared__ float red[8];
  const float* w = blockIdx.y==0 ? w0 : blockIdx.y==1 ? w1 : blockIdx.y==2 ? w2 : w3;
  const int stride = gridDim.x * 256;
  float s = 0.f;
  for (int i = blockIdx.x*256 + threadIdx.x; i < W_ELEMS; i += stride)
    s += fabsf(w[i]);
#pragma unroll
  for (int m=1;m<32;m<<=1) s += __shfl_xor(s,m,32);
  if ((threadIdx.x & 31) == 0) red[threadIdx.x>>5] = s;
  __syncthreads();
  if (threadIdx.x == 0){
    float t = 0.f;
#pragma unroll
    for (int i=0;i<8;++i) t += red[i];
    atomicAdd(&acc[blockIdx.y], t);
  }
}

__global__ void __launch_bounds__(256)
w_quant_kernel(const float* __restrict__ w0, const float* __restrict__ w1,
               const float* __restrict__ w2, const float* __restrict__ w3,
               const float* __restrict__ acc, int8_t* __restrict__ wq8){
  const int wi = blockIdx.y;
  const float* w = wi==0 ? w0 : wi==1 ? w1 : wi==2 ? w2 : w3;
  const float gamma = fmaxf(acc[wi]*(1.0f/(float)W_ELEMS), 1e-5f);
  const float inv = 1.0f/gamma;
  const int i = blockIdx.x*256 + threadIdx.x;
  const float q = fminf(fmaxf(rintf(w[i]*inv), -1.f), 1.f);
  wq8[(size_t)wi*W_ELEMS + i] = (int8_t)q;
}

__global__ void __launch_bounds__(256)
act_quant_kernel(const float* __restrict__ x, int8_t* __restrict__ xq,
                 float* __restrict__ xs){
  __shared__ float red[8];
  const int row = blockIdx.x;
  const float* xr = x + (size_t)row*D_MODEL;
  float v[4]; float m = 0.f;
#pragma unroll
  for (int i=0;i<4;++i){ v[i] = xr[threadIdx.x + i*256]; m = fmaxf(m, fabsf(v[i])); }
#pragma unroll
  for (int msk=1;msk<32;msk<<=1) m = fmaxf(m, __shfl_xor(m,msk,32));
  if ((threadIdx.x & 31) == 0) red[threadIdx.x>>5] = m;
  __syncthreads();
  float mm = red[0];
#pragma unroll
  for (int i=1;i<8;++i) mm = fmaxf(mm, red[i]);
  const float gamma = fmaxf(mm, 1e-5f);
  const float s = 127.0f/gamma;
#pragma unroll
  for (int i=0;i<4;++i){
    float q = fminf(fmaxf(rintf(v[i]*s), -128.f), 127.f);
    xq[(size_t)row*D_MODEL + threadIdx.x + i*256] = (int8_t)q;
  }
  if (threadIdx.x == 0) xs[row] = gamma*(1.0f/127.0f);
}

__global__ void __launch_bounds__(256)
bit_gemm_iu8_kernel(const int8_t* __restrict__ xq, const float* __restrict__ xs,
                    const int8_t* __restrict__ w8, const float* __restrict__ wacc,
                    int wi, const float* __restrict__ bias,
                    void* __restrict__ dst, int mode){
  const int lane = threadIdx.x & 31;
  const int wave = threadIdx.x >> 5;
  const int half = lane >> 4;
  const int lr   = lane & 15;
  const int t0 = blockIdx.x*32 + (wave>>2)*16;
  const int n0 = blockIdx.y*64 + (wave&3)*16;
  const float wg = fmaxf(wacc[wi]*(1.0f/(float)W_ELEMS), 1e-5f);
  const int8_t* arow = xq + (size_t)(t0+lr)*D_MODEL;
  const int8_t* brow = w8 + (size_t)(n0+lr)*D_MODEL;
  v8i acc = zero8i();
#pragma unroll 4
  for (int k0=0;k0<D_MODEL;k0+=64){
    v8i a = load_a8(arow+k0, half);
    v8i b = load_b8(brow+k0, half);
    acc = __builtin_amdgcn_wmma_i32_16x16x64_iu8(true, a, true, b, acc, false, false);
  }
  const int n = n0 + lr;
  const float bn = bias[n];
#pragma unroll
  for (int r=0;r<8;++r){
    const int t = t0 + r + half*8;
    const float y = (float)acc[r] * (xs[t]*wg) + bn;
    if (mode == 0){
      ((float*)dst)[(size_t)t*D_MODEL + n] = y;
    } else {
      const int b_ = t>>11, tt = t&2047, h = n>>6, d = n&63;
      if (mode == 1)
        ((_Float16*)dst)[((size_t)(b_*NH+h)*SEQ_T + tt)*DH + d] = (_Float16)y;
      else
        ((_Float16*)dst)[((size_t)(b_*NH+h)*DH + d)*SEQ_T + tt] = (_Float16)y;
    }
  }
}

__global__ void __launch_bounds__(128)
bit_flash_attn_kernel(const _Float16* __restrict__ qf, const _Float16* __restrict__ kf,
                      const _Float16* __restrict__ vf, float* __restrict__ out,
                      float* __restrict__ ent){
  __shared__ _Float16 kbuf[4][2][32*DH];
  __shared__ _Float16 vbuf[4][2][DH*32];
  __shared__ _Float16 plds[4][512];
  const int lane = threadIdx.x & 31;
  const int wave = threadIdx.x >> 5;
  const int half = lane >> 4;
  const int lr   = lane & 15;
  const int bh   = blockIdx.x;
  const int qt0  = blockIdx.y*64 + wave*16;

  const uint32_t kds0 = lds_off(&kbuf[wave][0][0]), kds1 = lds_off(&kbuf[wave][1][0]);
  const uint32_t vds0 = lds_off(&vbuf[wave][0][0]), vds1 = lds_off(&vbuf[wave][1][0]);
  const uint64_t kgbase = (uint64_t)(uintptr_t)(kf + (size_t)bh*SEQ_T*DH);
  const uint64_t vgbase = (uint64_t)(uintptr_t)(vf + (size_t)bh*DH*SEQ_T);
  const int vr = (lane>>2), vc = (lane&3)*16;

  auto issue_chunk = [&](int kt0, uint32_t kdst, uint32_t vdst){
    const uint64_t ksrc = kgbase + (uint64_t)kt0*(DH*2) + (uint32_t)lane*16u;
#pragma unroll
    for (int i=0;i<8;++i)
      async_copy_b128(kdst + (uint32_t)(lane*16 + i*512), ksrc + (uint32_t)(i*512));
    const uint64_t vsrc = vgbase + (uint64_t)kt0*2u;
#pragma unroll
    for (int i=0;i<8;++i)
      async_copy_b128(vdst + (uint32_t)((vr + i*8)*64 + vc),
                      vsrc + (uint32_t)((vr + i*8)*(SEQ_T*2) + vc));
  };

  const _Float16* qrow = qf + ((size_t)bh*SEQ_T + qt0 + lr)*DH;
  const v16h qa0 = load_h16f(qrow,      half);
  const v16h qa1 = load_h16f(qrow + 32, half);

  float Mx[8], S[8], Tt[8];
  v8f O[4];
#pragma unroll
  for (int r=0;r<8;++r){ Mx[r] = -1e30f; S[r] = 0.f; Tt[r] = 0.f; }
#pragma unroll
  for (int dt=0;dt<4;++dt) O[dt] = zero8f();
  _Float16* pl = &plds[wave][0];

  issue_chunk(0, kds0, vds0);
  for (int c=0; c<SEQ_T/32; ++c){
    const int cur = c & 1;
    if (c+1 < SEQ_T/32){
      issue_chunk((c+1)*32, cur ? kds0 : kds1, cur ? vds0 : vds1);
      asm volatile("s_wait_asynccnt 0x10" ::: "memory");
    } else {
      asm volatile("s_wait_asynccnt 0x0" ::: "memory");
    }
    const _Float16* kb = &kbuf[wave][cur][0];
    const v16h kb00 = load_h16f(kb + lr*DH,            half);
    const v16h kb01 = load_h16f(kb + lr*DH + 32,       half);
    const v16h kb10 = load_h16f(kb + (16+lr)*DH,       half);
    const v16h kb11 = load_h16f(kb + (16+lr)*DH + 32,  half);
    v8f s0 = zero8f(), s1 = zero8f();
    s0 = __builtin_amdgcn_wmma_f32_16x16x32_f16(false,qa0,false,kb00,(short)0,s0,false,false);
    s0 = __builtin_amdgcn_wmma_f32_16x16x32_f16(false,qa1,false,kb01,(short)0,s0,false,false);
    s1 = __builtin_amdgcn_wmma_f32_16x16x32_f16(false,qa0,false,kb10,(short)0,s1,false,false);
    s1 = __builtin_amdgcn_wmma_f32_16x16x32_f16(false,qa1,false,kb11,(short)0,s1,false,false);

    float alpha[8], e0[8], e1[8];
#pragma unroll
    for (int r=0;r<8;++r){
      const float a = s0[r]*ATT_SCALE, b = s1[r]*ATT_SCALE;
      const float mx   = redmax16(fmaxf(a,b));
      const float mnew = fmaxf(Mx[r], mx);
      const float al   = __expf(Mx[r]-mnew);
      const float ea   = __expf(a-mnew), eb = __expf(b-mnew);
      const float sc = redsum16(ea+eb);
      const float tc = redsum16(ea*(a-mnew) + eb*(b-mnew));
      Tt[r] = al*(Tt[r] + (Mx[r]-mnew)*S[r]) + tc;
      S[r]  = al*S[r] + sc;
      Mx[r] = mnew;
      alpha[r] = al; e0[r] = ea; e1[r] = eb;
    }
#pragma unroll
    for (int dt=0;dt<4;++dt)
#pragma unroll
      for (int r=0;r<8;++r) O[dt][r] *= alpha[r];

#pragma unroll
    for (int r=0;r<8;++r){
      const int m = r + half*8;
      pl[m*32 + lr]      = (_Float16)e0[r];
      pl[m*32 + 16 + lr] = (_Float16)e1[r];
    }
    asm volatile("s_wait_dscnt 0" ::: "memory");
    const v16h pa = load_h16f(pl + lr*32, half);

    const _Float16* vb_ = &vbuf[wave][cur][0];
#pragma unroll
    for (int dt=0;dt<4;++dt){
      const v16h vb = load_h16f(vb_ + (dt*16 + lr)*32, half);
      O[dt] = __builtin_amdgcn_wmma_f32_16x16x32_f16(false,pa,false,vb,(short)0,O[dt],false,false);
    }
  }

  const int b_ = bh >> 4, h = bh & 15;
#pragma unroll
  for (int dt=0;dt<4;++dt)
#pragma unroll
    for (int r=0;r<8;++r){
      const int q = qt0 + r + half*8;
      const int col = h*DH + dt*16 + lr;
      out[((size_t)(b_*SEQ_T + q))*D_MODEL + col] = O[dt][r]/S[r];
    }

  float part = 0.f;
#pragma unroll
  for (int r=0;r<8;++r) part += __logf(S[r]) - Tt[r]/S[r];
  if (lr == 0) atomicAdd(ent, part * (1.0f/65536.0f));
}

extern "C" void kernel_launch(void* const* d_in, const int* in_sizes, int n_in,
                              void* d_out, int out_size, void* d_ws, size_t ws_size,
                              hipStream_t stream) {
  (void)in_sizes; (void)n_in; (void)out_size; (void)ws_size;
  const float* x  = (const float*)d_in[0];
  const float* wq = (const float*)d_in[1];
  const float* bq = (const float*)d_in[2];
  const float* wk = (const float*)d_in[3];
  const float* bk = (const float*)d_in[4];
  const float* wv = (const float*)d_in[5];
  const float* bv = (const float*)d_in[6];
  const float* wo = (const float*)d_in[7];
  const float* bo = (const float*)d_in[8];

  char* ws = (char*)d_ws;
  constexpr size_t OFF_XQ  = 0;
  constexpr size_t OFF_XS  = OFF_XQ  + 4194304;
  constexpr size_t OFF_W8  = OFF_XS  + 16384;
  constexpr size_t OFF_ACC = OFF_W8  + 4194304;
  constexpr size_t OFF_QF  = OFF_ACC + 256;
  constexpr size_t OFF_KF  = OFF_QF  + 8388608;
  constexpr size_t OFF_VF  = OFF_KF  + 8388608;
  constexpr size_t OFF_AO  = OFF_VF  + 8388608;
  constexpr size_t OFF_XQ2 = OFF_AO  + 16777216;
  constexpr size_t OFF_XS2 = OFF_XQ2 + 4194304;

  int8_t*    xq_a = (int8_t*)   (ws + OFF_XQ);
  float*     xs_a = (float*)    (ws + OFF_XS);
  int8_t*    w8   = (int8_t*)   (ws + OFF_W8);
  float*     acc  = (float*)    (ws + OFF_ACC);
  _Float16*  qfp  = (_Float16*) (ws + OFF_QF);
  _Float16*  kfp  = (_Float16*) (ws + OFF_KF);
  _Float16*  vfp  = (_Float16*) (ws + OFF_VF);
  float*     ao   = (float*)    (ws + OFF_AO);
  int8_t*    xq_o = (int8_t*)   (ws + OFF_XQ2);
  float*     xs_o = (float*)    (ws + OFF_XS2);

  float* out = (float*)d_out;
  float* ent = out + (size_t)N_TOK*D_MODEL;

  init_zero_kernel<<<1, 32, 0, stream>>>(acc, ent);
  wabs_sum_kernel<<<dim3(64,4), 256, 0, stream>>>(wq, wk, wv, wo, acc);
  w_quant_kernel<<<dim3(W_ELEMS/256,4), 256, 0, stream>>>(wq, wk, wv, wo, acc, w8);
  act_quant_kernel<<<N_TOK, 256, 0, stream>>>(x, xq_a, xs_a);

  const dim3 ggrid(N_TOK/32, D_MODEL/64);
  bit_gemm_iu8_kernel<<<ggrid, 256, 0, stream>>>(xq_a, xs_a, w8,             acc, 0, bq, (void*)qfp, 1);
  bit_gemm_iu8_kernel<<<ggrid, 256, 0, stream>>>(xq_a, xs_a, w8 +   W_ELEMS, acc, 1, bk, (void*)kfp, 1);
  bit_gemm_iu8_kernel<<<ggrid, 256, 0, stream>>>(xq_a, xs_a, w8 + 2*W_ELEMS, acc, 2, bv, (void*)vfp, 2);

  bit_flash_attn_kernel<<<dim3(32, SEQ_T/64), 128, 0, stream>>>(qfp, kfp, vfp, ao, ent);

  act_quant_kernel<<<N_TOK, 256, 0, stream>>>(ao, xq_o, xs_o);
  bit_gemm_iu8_kernel<<<ggrid, 256, 0, stream>>>(xq_o, xs_o, w8 + 3*W_ELEMS, acc, 3, bo, (void*)out, 0);
}
